// PositionWiseFeedForward_34918084117133
// MI455X (gfx1250) — matrix operand order repaired and checked by reading, not compiled, not run on hardware
//
#include <hip/hip_runtime.h>
#include <stdint.h>

// Problem constants (from reference): B=4, S=2048, D=1024, F=4096, E=8, K=2
#define B_    4
#define S_    2048
#define D_    1024
#define F_    4096
#define E_    8
#define TOPK_ 2
#define T_    (B_ * S_)          // 8192 tokens
#define NSLOT (T_ * TOPK_)       // 16384 (token, k) slots -- fixed
#define TILE_M 128
#define MAXSLOT (NSLOT + E_ * TILE_M)   // 17408 after per-expert padding
#define CH     4096                      // slots per h-chunk (h chunk = 32 MB bf16)
#define NCHUNK ((MAXSLOT + CH - 1) / CH) // 5

typedef __attribute__((ext_vector_type(16))) __bf16 v16bf;
typedef __attribute__((ext_vector_type(8)))  float  v8f;

union FragAB { v16bf v; uint4 q[2]; };

__device__ __forceinline__ unsigned short f2bf(float f) {
  // round-to-nearest-even fp32 -> bf16
  unsigned int u = __builtin_bit_cast(unsigned int, f);
  u += 0x7FFFu + ((u >> 16) & 1u);
  return (unsigned short)(u >> 16);
}
__device__ __forceinline__ unsigned int pk2bf(float lo, float hi) {
  return (unsigned int)f2bf(lo) | ((unsigned int)f2bf(hi) << 16);
}

// ---------------------------------------------------------------- zero output
__global__ void moe_zero_kernel(float4* __restrict__ out, int n4) {
  int i = blockIdx.x * blockDim.x + threadIdx.x;
  if (i < n4) out[i] = make_float4(0.f, 0.f, 0.f, 0.f);
}

// ---------------------------------------------------------------- router
// One block per token: logits = x[t] @ gate_w + gate_b ; softmax ; top-2.
__global__ __launch_bounds__(256) void moe_router_kernel(
    const float* __restrict__ x, const float* __restrict__ gw,
    const float* __restrict__ gb, int* __restrict__ tok_e,
    float* __restrict__ tok_w) {
  __shared__ float red[256];
  __shared__ float logit_s[E_];
  int t = blockIdx.x;
  int tid = threadIdx.x;
  float part[E_];
#pragma unroll
  for (int e = 0; e < E_; ++e) part[e] = 0.f;
  const float* xr = x + (size_t)t * D_;
  for (int d = tid; d < D_; d += 256) {
    float xv = xr[d];
    const float* g = gw + (size_t)d * E_;  // gate_w is [D,E] row-major
#pragma unroll
    for (int e = 0; e < E_; ++e) part[e] += xv * g[e];
  }
  for (int e = 0; e < E_; ++e) {           // deterministic tree reductions
    red[tid] = part[e];
    __syncthreads();
    for (int s = 128; s > 0; s >>= 1) {
      if (tid < s) red[tid] += red[tid + s];
      __syncthreads();
    }
    if (tid == 0) logit_s[e] = red[0] + gb[e];
    __syncthreads();
  }
  if (tid == 0) {
    float p[E_];
    float m = -3.4e38f;
    for (int e = 0; e < E_; ++e) m = fmaxf(m, logit_s[e]);
    float s = 0.f;
    for (int e = 0; e < E_; ++e) { p[e] = __expf(logit_s[e] - m); s += p[e]; }
    float inv = 1.f / s;
    for (int e = 0; e < E_; ++e) p[e] *= inv;
    // top-2, ties -> lower index (matches jax.lax.top_k)
    int i1 = 0;
    for (int e = 1; e < E_; ++e) if (p[e] > p[i1]) i1 = e;
    int i2 = (i1 == 0) ? 1 : 0;
    for (int e = 0; e < E_; ++e) { if (e == i1) continue; if (p[e] > p[i2]) i2 = e; }
    tok_e[2 * t]     = i1; tok_w[2 * t]     = p[i1];
    tok_e[2 * t + 1] = i2; tok_w[2 * t + 1] = p[i2];
  }
}

// ---------------------------------------------------------------- compaction
// Group the 16384 slots by expert; pad each expert region to a multiple of
// TILE_M with weight-0 dummies so every GEMM tile is full and single-expert.
__global__ __launch_bounds__(256) void moe_compact_kernel(
    const int* __restrict__ tok_e, const float* __restrict__ tok_w,
    int* __restrict__ slot_tok, float* __restrict__ slot_w,
    int* __restrict__ slot_e, int* __restrict__ meta) {
  __shared__ int cnt[E_], cur[E_], off[E_ + 1];
  int tid = threadIdx.x;
  if (tid < E_) cnt[tid] = 0;
  __syncthreads();
  for (int i = tid; i < NSLOT; i += 256) atomicAdd(&cnt[tok_e[i]], 1);
  __syncthreads();
  if (tid == 0) {
    int start = 0;
    for (int e = 0; e < E_; ++e) {
      off[e] = start;
      cur[e] = start;
      start += ((cnt[e] + TILE_M - 1) / TILE_M) * TILE_M;
    }
    off[E_] = start;
    meta[0] = start;  // paddedTotal
  }
  __syncthreads();
  for (int i = tid; i < NSLOT; i += 256) {
    int e = tok_e[i];
    int pos = atomicAdd(&cur[e], 1);   // order within expert is irrelevant:
    slot_tok[pos] = i >> 1;            // GEMM rows are position-independent
    slot_w[pos] = tok_w[i];
    slot_e[pos] = e;
  }
  __syncthreads();
  int total = off[E_];
  for (int p = tid; p < total; p += 256) {
    int e = 0;
    while (p >= off[e + 1]) ++e;
    if (p >= cur[e]) {  // cur[e] == off[e] + cnt[e] now -> padding region
      slot_tok[p] = 0;
      slot_w[p] = 0.f;
      slot_e[p] = e;
    }
  }
}

// ---------------------------------------------------------------- GEMM1
// h[slot, f] = silu(x[token(slot)] @ W1[e] + b1[e]), bf16 WMMA, fp32 accum.
// Block tile 128x128, 8 waves in 4x2, each wave 32x64 (8x v_wmma 16x16x32).
__global__ __launch_bounds__(256) void moe_gemm1_kernel(
    const float* __restrict__ x, const float* __restrict__ w1,
    const float* __restrict__ b1, const int* __restrict__ slot_tok,
    const int* __restrict__ slot_e, const int* __restrict__ meta,
    unsigned short* __restrict__ hbuf, int chunkStart) {
  __shared__ int sTok[TILE_M];
  __shared__ __align__(16) unsigned short sA[TILE_M * 40];  // [128][32] +pad
  __shared__ __align__(16) unsigned short sB[128 * 40];     // B^T [128][32] +pad

  int paddedTotal = meta[0];
  int slotBase = chunkStart + blockIdx.y * TILE_M;
  if (slotBase >= paddedTotal) return;
  int expert = slot_e[slotBase];
  const float* w1e = w1 + (size_t)expert * D_ * F_;
  const float* b1e = b1 + (size_t)expert * F_;
  int f0 = blockIdx.x * 128;
  int tid = threadIdx.x;

  if (tid < TILE_M) sTok[tid] = slot_tok[slotBase + tid];
  __syncthreads();

  int lane = tid & 31, half = lane >> 4, n16 = lane & 15;
  int wave = tid >> 5, wm = wave >> 1, wn = wave & 1;
  unsigned int* sA32 = (unsigned int*)sA;

  v8f zero = {};
  v8f acc[2][4];
#pragma unroll
  for (int i = 0; i < 2; ++i)
#pragma unroll
    for (int j = 0; j < 4; ++j) acc[i][j] = zero;

  for (int k0 = 0; k0 < D_; k0 += 32) {
    // stage A: gathered x rows, float4 loads -> packed bf16 b64 LDS stores
    for (int u = tid; u < TILE_M * 8; u += 256) {  // 8 float4 per 32-wide row
      int row = u >> 3, c4 = u & 7;
      float4 v = *(const float4*)&x[(size_t)sTok[row] * D_ + k0 + c4 * 4];
      uint2 pk = make_uint2(pk2bf(v.x, v.y), pk2bf(v.z, v.w));
      *(uint2*)&sA32[row * 20 + c4 * 2] = pk;
    }
    // stage B: W1 tile [32 x 128], float4 loads, transposed into [128][32]
    for (int u = tid; u < 32 * 32; u += 256) {  // 32 float4 per 128-wide row
      int k = u >> 5, f4 = u & 31;
      float4 v = *(const float4*)&w1e[(size_t)(k0 + k) * F_ + f0 + f4 * 4];
      sB[(f4 * 4 + 0) * 40 + k] = f2bf(v.x);
      sB[(f4 * 4 + 1) * 40 + k] = f2bf(v.y);
      sB[(f4 * 4 + 2) * 40 + k] = f2bf(v.z);
      sB[(f4 * 4 + 3) * 40 + k] = f2bf(v.w);
    }
    __syncthreads();
    FragAB a[2], b[4];
#pragma unroll
    for (int sm = 0; sm < 2; ++sm) {  // A lane layout: K = 8*half+0..7, 16+8*half+0..7
      const unsigned short* base = &sA[(wm * 32 + sm * 16 + n16) * 40];
      a[sm].q[0] = *(const uint4*)(base + 8 * half);
      a[sm].q[1] = *(const uint4*)(base + 16 + 8 * half);
    }
#pragma unroll
    for (int sn = 0; sn < 4; ++sn) {  // B lane layout: K = 8*half+0..7, 16+8*half+0..7
      const unsigned short* base = &sB[(wn * 64 + sn * 16 + n16) * 40];
      b[sn].q[0] = *(const uint4*)(base + 8 * half);
      b[sn].q[1] = *(const uint4*)(base + 16 + 8 * half);
    }
#pragma unroll
    for (int sm = 0; sm < 2; ++sm)
#pragma unroll
      for (int sn = 0; sn < 4; ++sn)
        acc[sm][sn] = __builtin_amdgcn_wmma_f32_16x16x32_bf16(
            false, a[sm].v, false, b[sn].v, (short)0, acc[sm][sn], false, false);
    __syncthreads();
  }

  // epilogue: bias + SiLU, store bf16 h. C layout: N=lane%16, M=8*half+r.
#pragma unroll
  for (int sn = 0; sn < 4; ++sn) {
    int f = f0 + wn * 64 + sn * 16 + n16;
    float bias = b1e[f];
#pragma unroll
    for (int sm = 0; sm < 2; ++sm) {
      int rbase = blockIdx.y * TILE_M + wm * 32 + sm * 16 + 8 * half;  // in chunk
#pragma unroll
      for (int r = 0; r < 8; ++r) {
        float v = acc[sm][sn][r] + bias;
        v = v / (1.f + __expf(-v));  // silu
        hbuf[(size_t)(rbase + r) * F_ + f] = f2bf(v);
      }
    }
  }
}

// ---------------------------------------------------------------- GEMM2
// y[slot, d] = h[slot] @ W2[e] + b2[e];  out[token] += y * weight (atomic).
// A-operand (h, already bf16) is staged with GLOBAL_LOAD_ASYNC_TO_LDS_B128
// (ASYNCcnt DMA path, CDNA5 ISA) -- no VGPR round-trip.
__global__ __launch_bounds__(256) void moe_gemm2_kernel(
    const unsigned short* __restrict__ hbuf, const float* __restrict__ w2,
    const float* __restrict__ b2, const int* __restrict__ slot_tok,
    const float* __restrict__ slot_w, const int* __restrict__ slot_e,
    const int* __restrict__ meta, float* __restrict__ out, int chunkStart) {
  __shared__ __align__(16) unsigned short sA[TILE_M * 40];  // [128][32] +pad
  __shared__ __align__(16) unsigned short sB[64 * 40];      // B^T [64][32] +pad

  int paddedTotal = meta[0];
  int slotBase = chunkStart + blockIdx.y * TILE_M;
  if (slotBase >= paddedTotal) return;
  int expert = slot_e[slotBase];
  const float* w2e = w2 + (size_t)expert * F_ * D_;
  const float* b2e = b2 + (size_t)expert * D_;
  int d0 = blockIdx.x * 64;
  int tid = threadIdx.x;
  int hbase = blockIdx.y * TILE_M;

  int lane = tid & 31, half = lane >> 4, n16 = lane & 15;
  int wave = tid >> 5, wm = wave >> 1, wn = wave & 1;
  unsigned int sAldsBase = (unsigned int)(size_t)&sA[0];  // LDS byte address

  v8f zero = {};
  v8f acc[2][2];
#pragma unroll
  for (int i = 0; i < 2; ++i)
#pragma unroll
    for (int j = 0; j < 2; ++j) acc[i][j] = zero;

  for (int k0 = 0; k0 < F_; k0 += 32) {
    // stage A: h rows (bf16) via async DMA to LDS, 16B per lane per op.
    // tile = 128 rows x 64B; LDS row stride 80B (16B-aligned).
#pragma unroll
    for (int it = 0; it < 2; ++it) {
      int u = it * 256 + tid;            // 512 x 16B units
      int row = u >> 2, c = u & 3;
      unsigned int ldsAddr = sAldsBase + (unsigned int)(row * 80 + c * 16);
      unsigned long long gaddr =
          (unsigned long long)(size_t)(hbuf + (size_t)(hbase + row) * F_ + k0) +
          (unsigned long long)(c * 16);
      asm volatile("global_load_async_to_lds_b128 %0, %1, off"
                   :: "v"(ldsAddr), "v"(gaddr)
                   : "memory");
    }
    // stage B: W2 tile [32 x 64], float4 loads, transposed into [64][32]
    for (int u = tid; u < 32 * 16; u += 256) {
      int k = u >> 4, d4 = u & 15;
      float4 v = *(const float4*)&w2e[(size_t)(k0 + k) * D_ + d0 + d4 * 4];
      sB[(d4 * 4 + 0) * 40 + k] = f2bf(v.x);
      sB[(d4 * 4 + 1) * 40 + k] = f2bf(v.y);
      sB[(d4 * 4 + 2) * 40 + k] = f2bf(v.z);
      sB[(d4 * 4 + 3) * 40 + k] = f2bf(v.w);
    }
    asm volatile("s_wait_asynccnt 0" ::: "memory");  // A-tile DMA landed
    __syncthreads();
    FragAB a[2], b[2];
#pragma unroll
    for (int sm = 0; sm < 2; ++sm) {
      const unsigned short* base = &sA[(wm * 32 + sm * 16 + n16) * 40];
      a[sm].q[0] = *(const uint4*)(base + 8 * half);
      a[sm].q[1] = *(const uint4*)(base + 16 + 8 * half);
    }
#pragma unroll
    for (int sn = 0; sn < 2; ++sn) {
      const unsigned short* base = &sB[(wn * 32 + sn * 16 + n16) * 40];
      b[sn].q[0] = *(const uint4*)(base + 8 * half);
      b[sn].q[1] = *(const uint4*)(base + 16 + 8 * half);
    }
#pragma unroll
    for (int sm = 0; sm < 2; ++sm)
#pragma unroll
      for (int sn = 0; sn < 2; ++sn)
        acc[sm][sn] = __builtin_amdgcn_wmma_f32_16x16x32_bf16(
            false, a[sm].v, false, b[sn].v, (short)0, acc[sm][sn], false, false);
    __syncthreads();
  }

  // epilogue: (acc + b2) * combine_weight -> atomicAdd into out.
  // Each (token,d) receives exactly two nonzero adds -> commutative ->
  // bitwise deterministic. Padding slots have weight 0.
#pragma unroll
  for (int sm = 0; sm < 2; ++sm) {
    int srow = slotBase + wm * 32 + sm * 16 + 8 * half;
#pragma unroll
    for (int sn = 0; sn < 2; ++sn) {
      int d = d0 + wn * 32 + sn * 16 + n16;
      float bias = b2e[d];
#pragma unroll
      for (int r = 0; r < 8; ++r) {
        int s = srow + r;
        float wgt = slot_w[s];
        int tok = slot_tok[s];
        float val = (acc[sm][sn][r] + bias) * wgt;
        atomicAdd(&out[(size_t)tok * D_ + d], val);
      }
    }
  }
}

// ---------------------------------------------------------------- launch
extern "C" void kernel_launch(void* const* d_in, const int* in_sizes, int n_in,
                              void* d_out, int out_size, void* d_ws,
                              size_t ws_size, hipStream_t stream) {
  (void)in_sizes; (void)n_in; (void)ws_size;
  const float* x  = (const float*)d_in[0];
  const float* gw = (const float*)d_in[1];
  const float* gb = (const float*)d_in[2];
  const float* w1 = (const float*)d_in[3];
  const float* b1 = (const float*)d_in[4];
  const float* w2 = (const float*)d_in[5];
  const float* b2 = (const float*)d_in[6];
  float* out = (float*)d_out;

  // workspace layout (~32.5 MB total)
  char* ws = (char*)d_ws;
  int*   meta     = (int*)(ws);
  int*   tok_e    = (int*)(ws + 64);
  float* tok_w    = (float*)(ws + 64 + 4 * (size_t)NSLOT);
  int*   slot_tok = (int*)(ws + 64 + 8 * (size_t)NSLOT);
  float* slot_w   = (float*)(ws + 64 + 8 * (size_t)NSLOT + 4 * (size_t)MAXSLOT);
  int*   slot_e   = (int*)(ws + 64 + 8 * (size_t)NSLOT + 8 * (size_t)MAXSLOT);
  size_t hoff = (64 + 8 * (size_t)NSLOT + 12 * (size_t)MAXSLOT + 255) & ~(size_t)255;
  unsigned short* hbuf = (unsigned short*)(ws + hoff);

  int n4 = out_size / 4;  // out_size = 8,388,608 floats (divisible by 4)
  moe_zero_kernel<<<(n4 + 255) / 256, 256, 0, stream>>>((float4*)out, n4);

  moe_router_kernel<<<T_, 256, 0, stream>>>(x, gw, gb, tok_e, tok_w);
  moe_compact_kernel<<<1, 256, 0, stream>>>(tok_e, tok_w, slot_tok, slot_w,
                                            slot_e, meta);

  for (int c = 0; c < NCHUNK; ++c) {
    int cs = c * CH;
    moe_gemm1_kernel<<<dim3(F_ / 128, CH / TILE_M), 256, 0, stream>>>(
        x, w1, b1, slot_tok, slot_e, meta, hbuf, cs);
    moe_gemm2_kernel<<<dim3(D_ / 64, CH / TILE_M), 256, 0, stream>>>(
        hbuf, w2, b2, slot_tok, slot_w, slot_e, meta, out, cs);
  }
}
